// BoundingBoxRegressorAndMultiLabelClassifier_v5_37237366456335
// MI455X (gfx1250) — matrix operand order repaired and checked by reading, not compiled, not run on hardware
//
#include <hip/hip_runtime.h>
#include <hip/hip_bf16.h>

typedef __bf16 bf16_t;
typedef __attribute__((ext_vector_type(4)))  __bf16 v4bf;
typedef __attribute__((ext_vector_type(8)))  __bf16 v8bf;
typedef __attribute__((ext_vector_type(16))) __bf16 v16bf;
typedef __attribute__((ext_vector_type(8)))  float  v8f;

#define B_      16
#define C_      512
#define H_      32
#define HW_     1024
#define HID_    512
#define NB_     36
#define S_      26
#define P_      7
#define IN_DIM_ 25088      // 7*7*512
#define M_      576        // B_*NB_
#define KDIM_   25088
#define GRP_SZ_ 6

#define CHUNK_K_ 128                     // K elements staged per LDS buffer
#define NCHUNK_  (KDIM_ / CHUNK_K_)      // 196
#define APAD_    136                     // padded LDS row stride (bf16 elems)

// ---------- bf16 helpers (bit-exact RNE, no reliance on __bf16 arithmetic) ----
__device__ __forceinline__ bf16_t f2bf(float f) {
    unsigned u = __builtin_bit_cast(unsigned, f);
    unsigned r = (u + 0x7FFFu + ((u >> 16) & 1u)) >> 16;
    return __builtin_bit_cast(bf16_t, (unsigned short)r);
}
__device__ __forceinline__ float bf2f(bf16_t b) {
    unsigned short s = __builtin_bit_cast(unsigned short, b);
    return __builtin_bit_cast(float, (unsigned)s << 16);
}
__device__ __forceinline__ v16bf cat8(v8bf lo, v8bf hi) {
    return __builtin_shufflevector(lo, hi, 0,1,2,3,4,5,6,7,8,9,10,11,12,13,14,15);
}

// ---------- 1. (B,C,H,W) f32 -> (B,HW,C) bf16, tiled LDS transpose ------------
__global__ __launch_bounds__(256)
void transpose_feat_kernel(const float* __restrict__ lf, bf16_t* __restrict__ featl) {
    __shared__ float tile[32][33];
    const int b  = blockIdx.z;
    const int p0 = blockIdx.x * 32;     // position tile (y*W+x)
    const int c0 = blockIdx.y * 32;     // channel tile
    const int tx = threadIdx.x, ty = threadIdx.y;
    #pragma unroll
    for (int i = 0; i < 4; ++i) {
        int c = c0 + ty + i * 8;
        tile[ty + i * 8][tx] = lf[((size_t)(b * C_ + c)) * HW_ + p0 + tx];
    }
    __syncthreads();
    #pragma unroll
    for (int i = 0; i < 4; ++i) {
        int p = p0 + ty + i * 8;
        featl[((size_t)(b * HW_ + p)) * C_ + c0 + tx] = f2bf(tile[tx][ty + i * 8]);
    }
}

// ---------- 2. roi_fc_W f32 -> bf16 ------------------------------------------
__global__ __launch_bounds__(256)
void convert_w_kernel(const float* __restrict__ w, bf16_t* __restrict__ wbf) {
    size_t i = (size_t)blockIdx.x * blockDim.x + threadIdx.x;   // quad index
    float4 f = ((const float4*)w)[i];
    v4bf r;
    r.x = f2bf(f.x); r.y = f2bf(f.y); r.z = f2bf(f.z); r.w = f2bf(f.w);
    ((v4bf*)wbf)[i] = r;
}

// ---------- 3. ROI align -> A [M_, IN_DIM_] bf16 -----------------------------
__global__ __launch_bounds__(256)
void roi_align_kernel(const bf16_t* __restrict__ featl,
                      const float* __restrict__ coords,   // (B,S,4)
                      bf16_t* __restrict__ A) {
    const int m = blockIdx.x;            // 0..575
    const int b = m / NB_, n = m % NB_;
    const int tid = threadIdx.x;

    float c0, c1, c2, c3;
    if (n < S_) {
        const float* p = coords + ((size_t)(b * S_ + n)) * 4;
        c0 = p[0]; c1 = p[1]; c2 = p[2]; c3 = p[3];
    } else { c0 = 0.f; c1 = 0.f; c2 = 1.f; c3 = 1.f; }
    const float x1 = c0 * 32.f, y1 = c1 * 32.f, x2 = c2 * 32.f, y2 = c3 * 32.f;
    const float rw = fmaxf(x2 - x1, 1.f), rh = fmaxf(y2 - y1, 1.f);
    const float bw = rw / (float)P_,     bh = rh / (float)P_;

    __shared__ float s_fr[2][14];   // frac
    __shared__ int   s_lo[2][14], s_hi[2][14], s_vd[2][14];
    if (tid < 28) {
        int axis = tid / 14, j = tid % 14;            // 0=y, 1=x
        float base = axis ? x1 : y1;
        float bsz  = axis ? bw : bh;
        float off  = (float)(j >> 1) + ((float)(j & 1) + 0.5f) * 0.5f;
        float t    = base + off * bsz;
        int   vd   = (t > -1.f) && (t < 32.f);
        float tc   = fminf(fmaxf(t, 0.f), 31.f);
        int   lo   = (int)floorf(tc);
        int   hi   = min(lo + 1, 31);
        s_fr[axis][j] = tc - (float)lo;
        s_lo[axis][j] = lo; s_hi[axis][j] = hi; s_vd[axis][j] = vd;
    }
    __syncthreads();

    const int ch0 = tid, ch1 = tid + 256;
    const size_t fb = (size_t)b * HW_ * C_;
    for (int cell = 0; cell < P_ * P_; ++cell) {
        const int py = cell / P_, px = cell % P_;
        float a0 = 0.f, a1 = 0.f;
        #pragma unroll
        for (int s = 0; s < 4; ++s) {
            int jy = py * 2 + (s >> 1), jx = px * 2 + (s & 1);
            if (s_vd[0][jy] && s_vd[1][jx]) {
                float ly = s_fr[0][jy], lx = s_fr[1][jx];
                float hy = 1.f - ly,    hx = 1.f - lx;
                int yl = s_lo[0][jy], yh = s_hi[0][jy];
                int xl = s_lo[1][jx], xh = s_hi[1][jx];
                float w00 = hy * hx, w01 = hy * lx, w10 = ly * hx, w11 = ly * lx;
                size_t i00 = fb + ((size_t)(yl * H_ + xl)) * C_;
                size_t i01 = fb + ((size_t)(yl * H_ + xh)) * C_;
                size_t i10 = fb + ((size_t)(yh * H_ + xl)) * C_;
                size_t i11 = fb + ((size_t)(yh * H_ + xh)) * C_;
                a0 += bf2f(featl[i00 + ch0]) * w00 + bf2f(featl[i01 + ch0]) * w01
                    + bf2f(featl[i10 + ch0]) * w10 + bf2f(featl[i11 + ch0]) * w11;
                a1 += bf2f(featl[i00 + ch1]) * w00 + bf2f(featl[i01 + ch1]) * w01
                    + bf2f(featl[i10 + ch1]) * w10 + bf2f(featl[i11 + ch1]) * w11;
            }
        }
        size_t arow = (size_t)m * IN_DIM_;
        A[arow + (size_t)ch0 * 49 + cell] = f2bf(a0 * 0.25f);
        A[arow + (size_t)ch1 * 49 + cell] = f2bf(a1 * 0.25f);
    }
}

// ---------- 4. WMMA GEMM: feats = relu(A[576,25088] x W[512,25088]^T + b) ----
// One block = one M tile (16 rows). The A chunk (16 x 128 bf16 = 4 KB) is
// staged in LDS once per block via global_load_async_to_lds_b128 (ASYNCcnt),
// double-buffered, and shared by all 8 waves; each wave owns a 64-wide N strip
// of 4 tiles (1 A-frag from LDS reused across 4 WMMAs, B direct from global).
__global__ __launch_bounds__(256)
void gemm_wmma_kernel(const bf16_t* __restrict__ Abf,
                      const bf16_t* __restrict__ Wbf,
                      const float*  __restrict__ bias,
                      float* __restrict__ feats) {
    __shared__ bf16_t Asmem[2][16][APAD_];     // 2 x 16 x 136 bf16 = 8.5 KB

    const int t     = threadIdx.x;
    const int lane  = t & 31;
    const int strip = t >> 5;                  // 0..7 (wave id = N strip)
    const int mt    = blockIdx.x;              // 0..35 (M tile)
    const int r = lane & 15, h = lane >> 4;

    // --- async copy setup: thread t moves one 16B slice per chunk ------------
    const int rowA = t >> 4;                   // 0..15 (A row in tile)
    const int colA = t & 15;                   // 0..15 (8-elem slice in chunk)
    const bf16_t* gslice = Abf + (size_t)(mt * 16 + rowA) * KDIM_ + colA * 8;

    // B fragment rows: lane = N col, K = 8h+0..7 and 16+8h+0..7 (ISA 7.12.2)
    const bf16_t* bRow[4];
    #pragma unroll
    for (int j = 0; j < 4; ++j)
        bRow[j] = Wbf + (size_t)((strip * 4 + j) * 16 + r) * KDIM_ + 8 * h;

    v8f acc[4] = {};

    // prologue: stage chunk 0
    {
        unsigned laddr = (unsigned)(uintptr_t)&Asmem[0][rowA][colA * 8];
        unsigned long long ga = (unsigned long long)(uintptr_t)gslice;
        asm volatile("global_load_async_to_lds_b128 %0, %1, off"
                     :: "v"(laddr), "v"(ga) : "memory");
    }

    for (int c = 0; c < NCHUNK_; ++c) {
        if (c + 1 < NCHUNK_) {
            // stage chunk c+1 into the other buffer
            unsigned laddr = (unsigned)(uintptr_t)&Asmem[(c + 1) & 1][rowA][colA * 8];
            unsigned long long ga =
                (unsigned long long)(uintptr_t)(gslice + (size_t)(c + 1) * CHUNK_K_);
            asm volatile("global_load_async_to_lds_b128 %0, %1, off"
                         :: "v"(laddr), "v"(ga) : "memory");
            asm volatile("s_wait_asynccnt 0x1" ::: "memory");  // chunk c landed (this wave)
        } else {
            asm volatile("s_wait_asynccnt 0x0" ::: "memory");
        }
        __syncthreads();                       // chunk c visible from all waves

        // A-frag (ISA 7.12.2 16-bit A layout): lane = M row; halves 0-7 ->
        // K = 8h+0..7, halves 8-15 -> K = 16+8h+0..7, read from LDS.
        const bf16_t* aRow = &Asmem[c & 1][r][8 * h];
        const size_t  kg   = (size_t)c * CHUNK_K_;
        #pragma unroll
        for (int s = 0; s < 4; ++s) {          // 4 k-steps of 32 per chunk
            v8bf a0 = *(const v8bf*)(aRow + s * 32);
            v8bf a1 = *(const v8bf*)(aRow + s * 32 + 16);
            v16bf afrag = cat8(a0, a1);
            #pragma unroll
            for (int j = 0; j < 4; ++j) {
                v8bf b0 = *(const v8bf*)(bRow[j] + kg + s * 32);
                v8bf b1 = *(const v8bf*)(bRow[j] + kg + s * 32 + 16);
                v16bf bfrag = cat8(b0, b1);
                acc[j] = __builtin_amdgcn_wmma_f32_16x16x32_bf16(
                    false, afrag, false, bfrag, (short)0, acc[j], false, false);
            }
        }
        __syncthreads();                       // done reading buf before re-stage
    }

    // C/D layout: VGPR i -> M = i + 8h, N = lane&15
    #pragma unroll
    for (int j = 0; j < 4; ++j) {
        int col = (strip * 4 + j) * 16 + r;
        float bcol = bias[col];
        #pragma unroll
        for (int i = 0; i < 8; ++i) {
            int row = mt * 16 + i + 8 * h;
            float v = acc[j][i] + bcol;
            feats[(size_t)row * HID_ + col] = v > 0.f ? v : 0.f;
        }
    }
}

// ---------- 5. Heads: one wave per output scalar -----------------------------
__global__ __launch_bounds__(256)
void heads_kernel(const float* __restrict__ feats,
                  const float* __restrict__ coords,   // (B,26,4)
                  const float* __restrict__ cW, const float* __restrict__ cB,
                  const float* __restrict__ pW, const float* __restrict__ pB,
                  const float* __restrict__ lW, const float* __restrict__ lB,
                  const float* __restrict__ gW, const float* __restrict__ gB,
                  const int*   __restrict__ locIdx,
                  const int*   __restrict__ grpIdx,
                  float* __restrict__ out) {
    const int o    = blockIdx.x * (blockDim.x >> 5) + (threadIdx.x >> 5);
    const int lane = threadIdx.x & 31;
    if (o >= 4640) return;

    float sum = 0.f;
    float result = 0.f;

    if (o < 1664) {                         // refined (B,26,4)
        int b = o / 104, rem = o % 104, s = rem / 4, j = rem % 4;
        const float* f = feats + (size_t)(b * NB_ + s) * HID_;
        const float* w = cW + (size_t)(s * 4 + j) * HID_;
        for (int k = lane; k < HID_; k += 32) sum += f[k] * w[k];
        #pragma unroll
        for (int off = 16; off; off >>= 1) sum += __shfl_xor(sum, off, 32);
        const float* cd = coords + (size_t)(b * S_ + s) * 4;
        float bw = cd[2] - cd[0], bh = cd[3] - cd[1];
        float wh = (j & 1) ? bh : bw;
        result = cd[j] + (sum + cB[s * 4 + j]) * wh;
    } else if (o < 2080) {                  // presence (B,26)
        int i = o - 1664, b = i / S_, s = i % S_;
        const float* f = feats + (size_t)(b * NB_ + s) * HID_;
        const float* w = pW + (size_t)s * HID_;
        for (int k = lane; k < HID_; k += 32) sum += f[k] * w[k];
        #pragma unroll
        for (int off = 16; off; off >>= 1) sum += __shfl_xor(sum, off, 32);
        result = sum + pB[s];
    } else {                                // mlc (B,160) = loc(96) | grp(64)
        int i = o - 2080, b = i / 160, rr = i % 160;
        if (rr < 96) {
            int l = rr / 8, lab = rr % 8;
            int n = locIdx[l];
            const float* f = feats + (size_t)(b * NB_ + n) * HID_;
            const float* w = lW + (size_t)(l * 8 + lab) * HID_;
            for (int k = lane; k < HID_; k += 32) sum += f[k] * w[k];
            #pragma unroll
            for (int off = 16; off; off >>= 1) sum += __shfl_xor(sum, off, 32);
            result = sum + lB[l * 8 + lab];
        } else {
            int q = rr - 96, g = q / 16, lab = q % 16;
            const float* w = gW + (size_t)(g * 16 + lab) * (GRP_SZ_ * HID_);
            for (int k = lane; k < GRP_SZ_ * HID_; k += 32) {
                int n = grpIdx[g * GRP_SZ_ + (k >> 9)];
                sum += feats[(size_t)(b * NB_ + n) * HID_ + (k & 511)] * w[k];
            }
            #pragma unroll
            for (int off = 16; off; off >>= 1) sum += __shfl_xor(sum, off, 32);
            result = sum + gB[g * 16 + lab];
        }
    }
    if (lane == 0) out[o] = result;
}

// ---------- launcher ----------------------------------------------------------
extern "C" void kernel_launch(void* const* d_in, const int* in_sizes, int n_in,
                              void* d_out, int out_size, void* d_ws, size_t ws_size,
                              hipStream_t stream) {
    const float* lf     = (const float*)d_in[0];
    const float* coords = (const float*)d_in[1];
    const float* roiW   = (const float*)d_in[2];
    const float* roiB   = (const float*)d_in[3];
    const float* cW     = (const float*)d_in[4];
    const float* cB     = (const float*)d_in[5];
    const float* pW     = (const float*)d_in[6];
    const float* pB     = (const float*)d_in[7];
    const float* lW     = (const float*)d_in[8];
    const float* lB     = (const float*)d_in[9];
    const float* gW     = (const float*)d_in[10];
    const float* gB     = (const float*)d_in[11];
    const int*   locIdx = (const int*)d_in[12];
    const int*   grpIdx = (const int*)d_in[13];
    float* out = (float*)d_out;

    // workspace layout (all 256B aligned)
    char* ws = (char*)d_ws;
    bf16_t* featl = (bf16_t*)(ws);                    // 16 MiB  (B*HW*C bf16)
    bf16_t* Wbf   = (bf16_t*)(ws + 16777216);         // 24.5 MiB (512*25088 bf16)
    bf16_t* Abf   = (bf16_t*)(ws + 42467328);         // 27.6 MiB (576*25088 bf16)
    float*  feats = (float*)(ws + 71368704);          // 1.1 MiB  (576*512 f32)

    transpose_feat_kernel<<<dim3(HW_ / 32, C_ / 32, B_), dim3(32, 8), 0, stream>>>(lf, featl);
    convert_w_kernel<<<(HID_ * IN_DIM_) / (256 * 4), 256, 0, stream>>>(roiW, Wbf);
    roi_align_kernel<<<M_, 256, 0, stream>>>(featl, coords, Abf);
    gemm_wmma_kernel<<<M_ / 16, 256, 0, stream>>>(Abf, Wbf, roiB, feats);
    heads_kernel<<<(4640 + 7) / 8, 256, 0, stream>>>(feats, coords, cW, cB, pW, pB,
                                                     lW, lB, gW, gB, locIdx, grpIdx, out);
}
